// CausalSelfAttention_8083128451674
// MI455X (gfx1250) — matrix operand order repaired and checked by reading, not compiled, not run on hardware
//
#include <hip/hip_runtime.h>
#include <math.h>

// ---------------------------------------------------------------------------
// CDNA5 (gfx1250) wave32 WMMA attention block.
//   D(16x16 f32) = A(16x32 bf16) x B(32x16 bf16) + C   via v_wmma_f32_16x16x32_bf16
// Fragment layouts per CDNA5 ISA 7.12.2 (wave32):
//   A: lane L -> row M = L&15; element e -> K = e + 8*(L>>4) + (e>=8 ? 8 : 0)
//   B: lane L -> col N = L&15; element e -> K = e + 8*(L>>4) + (e>=8 ? 8 : 0)
//   C/D: lane L -> col N = L&15; vgpr r -> row M = r + 8*(L>>4)
// All LDS tiles are laid out so fragment reads are two contiguous 8-element runs
// (two ds_load_b128 per fragment); pads keep 16-byte alignment + bank spread.
// GEMM is templated on K/N so strides are compile-time constants and stores
// fold into immediate-offset global_store_b32.
// ---------------------------------------------------------------------------

typedef __attribute__((ext_vector_type(16))) __bf16 v16bf;
typedef __attribute__((ext_vector_type(8)))  float  v8f;

union V8F  { v8f  v; float  f[8];  };
union V16B { v16bf v; __bf16 b[16]; };

constexpr int B_  = 2;
constexpr int T_  = 2048;
constexpr int C_  = 2048;
constexpr int NH_ = 16;
constexpr int NKV_ = 4;
constexpr int HD_ = 128;
constexpr int GC_ = 32;

// ---------------------------------------------------------------------------
// f32 -> bf16 elementwise conversion
// ---------------------------------------------------------------------------
__global__ void cvt_f32_bf16(const float* __restrict__ in, __bf16* __restrict__ out, int n) {
    int i = blockIdx.x * blockDim.x + threadIdx.x;
    if (i < n) out[i] = (__bf16)in[i];
}

// ---------------------------------------------------------------------------
// Generic bf16 WMMA GEMM:  C(MxN,f32) = A(MxK,bf16) x B(KxN,bf16)
// Block: 128 threads (4 waves). Block tile 64x64, wave tile 32x32, K-step 32.
// As is row-major [m][k]; Bs is stored TRANSPOSED [n][k] (pad 8 -> stride 40,
// 80-byte rows: 16B-aligned fragment runs) so B fragments are contiguous.
// K, N are compile-time so all strides fold into immediate offsets.
// ---------------------------------------------------------------------------
template <int K, int N>
__global__ __launch_bounds__(128)
void gemm_bf16_wmma(const __bf16* __restrict__ A, const __bf16* __restrict__ Bm,
                    float* __restrict__ Cm) {
    __shared__ __bf16 As[64][32];
    __shared__ __bf16 Bs[64][40];     // [n][k] transposed, padded

    const int tid  = threadIdx.x;
    const int lane = tid & 31;
    const int wave = tid >> 5;
    const int half = lane >> 4;
    const int ml   = lane & 15;

    const int m0 = blockIdx.y * 64;
    const int n0 = blockIdx.x * 64;
    const int wm = (wave >> 1) * 32;
    const int wn = (wave & 1) * 32;

    V8F acc[2][2];
#pragma unroll
    for (int i = 0; i < 2; ++i)
#pragma unroll
        for (int j = 0; j < 2; ++j)
#pragma unroll
            for (int r = 0; r < 8; ++r) acc[i][j].f[r] = 0.0f;

    // Per-thread staging pointers (advance by constant stride each K step).
    const int arow = tid >> 1, acb = (tid & 1) * 16;
    const int brow = tid >> 2, bcb = (tid & 3) * 16;
    const __bf16* asrc = A  + (size_t)(m0 + arow) * K + acb;
    const __bf16* bsrc = Bm + (size_t)brow * N + n0 + bcb;

    for (int k0 = 0; k0 < K; k0 += 32) {
        // Stage A tile (64x32, row-major, contiguous ds_store_b128).
        __builtin_prefetch(asrc + 32, 0, 1);          // global_prefetch next K tile
#pragma unroll
        for (int j = 0; j < 16; ++j) As[arow][acb + j] = asrc[j];
        asrc += 32;

        // Stage B tile (32x64 in memory) transposed into Bs[n][k].
        __builtin_prefetch(bsrc + (size_t)32 * N, 0, 1);
#pragma unroll
        for (int j = 0; j < 16; ++j) Bs[bcb + j][brow] = bsrc[j];
        bsrc += (size_t)32 * N;

        __syncthreads();

        V16B afr[2], bfr[2];
#pragma unroll
        for (int i = 0; i < 2; ++i) {
            const int row = wm + i * 16 + ml;
#pragma unroll
            for (int e = 0; e < 8; ++e) {
                afr[i].b[e]     = As[row][8 * half + e];
                afr[i].b[e + 8] = As[row][16 + 8 * half + e];
            }
        }
#pragma unroll
        for (int j = 0; j < 2; ++j) {
            const int col = wn + j * 16 + ml;
#pragma unroll
            for (int e = 0; e < 16; ++e) bfr[j].b[e] = Bs[col][8 * half + e + (e >= 8 ? 8 : 0)];
        }

#pragma unroll
        for (int i = 0; i < 2; ++i)
#pragma unroll
            for (int j = 0; j < 2; ++j)
                acc[i][j].v = __builtin_amdgcn_wmma_f32_16x16x32_bf16(
                    false, afr[i].v, false, bfr[j].v,
                    (short)0, acc[i][j].v, false, false);
        __syncthreads();
    }

    // Epilogue: one per-thread base pointer, all else constant offsets.
    float* cbase = Cm + (size_t)(m0 + wm + 8 * half) * N + n0 + wn + ml;
#pragma unroll
    for (int i = 0; i < 2; ++i)
#pragma unroll
        for (int j = 0; j < 2; ++j)
#pragma unroll
            for (int r = 0; r < 8; ++r)
                cbase[(size_t)(i * 16 + r) * N + j * 16] = acc[i][j].f[r];
}

// ---------------------------------------------------------------------------
// Fused: gate = 2*sigmoid(x[:, :32] @ Wgate);  v += gate * ve;
// RoPE + RMS on q and k; emit bf16 q/k/v in [b, head, t, hd] layout.
// One block per (b, t), 128 threads (one per hd element).
// ---------------------------------------------------------------------------
__global__ __launch_bounds__(128)
void postqkv(const float* __restrict__ x,  const float* __restrict__ ve,
             const float* __restrict__ cosv, const float* __restrict__ sinv,
             const float* __restrict__ Wgate,
             const float* __restrict__ qf, const float* __restrict__ kf,
             const float* __restrict__ vf,
             __bf16* __restrict__ qb, __bf16* __restrict__ kb, __bf16* __restrict__ vb) {
    __shared__ float red[128];
    __shared__ float sgate[NKV_];

    const int bt  = blockIdx.x;
    const int b   = bt / T_;
    const int t   = bt % T_;
    const int tid = threadIdx.x;

    if (tid < NKV_) {
        float acc = 0.0f;
        for (int g = 0; g < GC_; ++g)
            acc += x[(size_t)bt * C_ + g] * Wgate[g * NKV_ + tid];
        sgate[tid] = 2.0f / (1.0f + __expf(-acc));
    }
    __syncthreads();

    const float c = cosv[(size_t)t * (HD_ / 2) + (tid & 63)];
    const float s = sinv[(size_t)t * (HD_ / 2) + (tid & 63)];
    const float eps = 1.1920929e-07f;

    // q heads: rope + rms
    for (int h = 0; h < NH_; ++h) {
        const float* qh = qf + ((size_t)bt * NH_ + h) * HD_;
        float r;
        if (tid < 64) { float x1 = qh[tid];      float x2 = qh[tid + 64]; r =  x1 * c + x2 * s; }
        else          { float x1 = qh[tid - 64]; float x2 = qh[tid];      r = -x1 * s + x2 * c; }
        red[tid] = r * r;
        __syncthreads();
        for (int off = 64; off > 0; off >>= 1) {
            if (tid < off) red[tid] += red[tid + off];
            __syncthreads();
        }
        const float scale = rsqrtf(red[0] / (float)HD_ + eps);
        __syncthreads();
        qb[(((size_t)b * NH_ + h) * T_ + t) * HD_ + tid] = (__bf16)(r * scale);
    }

    // k heads: rope + rms
    for (int h = 0; h < NKV_; ++h) {
        const float* kh = kf + ((size_t)bt * NKV_ + h) * HD_;
        float r;
        if (tid < 64) { float x1 = kh[tid];      float x2 = kh[tid + 64]; r =  x1 * c + x2 * s; }
        else          { float x1 = kh[tid - 64]; float x2 = kh[tid];      r = -x1 * s + x2 * c; }
        red[tid] = r * r;
        __syncthreads();
        for (int off = 64; off > 0; off >>= 1) {
            if (tid < off) red[tid] += red[tid + off];
            __syncthreads();
        }
        const float scale = rsqrtf(red[0] / (float)HD_ + eps);
        __syncthreads();
        kb[(((size_t)b * NKV_ + h) * T_ + t) * HD_ + tid] = (__bf16)(r * scale);
    }

    // v heads: gated value-embedding mix
    for (int kv = 0; kv < NKV_; ++kv) {
        const size_t idx = ((size_t)bt * NKV_ + kv) * HD_ + tid;
        const float vv = vf[idx] + sgate[kv] * ve[idx];
        vb[(((size_t)b * NKV_ + kv) * T_ + t) * HD_ + tid] = (__bf16)vv;
    }
}

// ---------------------------------------------------------------------------
// Causal GQA flash attention. One block = 128 threads = 4 waves.
// Wave w owns a 16-row query tile; block covers 64 q rows of one (b, h).
// All 4 waves share the same kv-head and the same key-block sequence, so the
// 32x128 K and V tiles are staged cooperatively into LDS once per step:
//   Ks[key][hd]   row-major, stride 136 (272B rows: aligned + bank-spread)
//   Vs[hd][key]   TRANSPOSED, stride 40 (80B rows) -> B-fragments contiguous
// All waves run a uniform number of steps (extra blocks fully masked) so
// __syncthreads() is structurally safe.
// ---------------------------------------------------------------------------
__global__ __launch_bounds__(128)
void attn_wmma(const __bf16* __restrict__ qb, const __bf16* __restrict__ kb,
               const __bf16* __restrict__ vb, __bf16* __restrict__ yb) {
    __shared__ __bf16 Ks[32][136];
    __shared__ __bf16 Vs[128][40];
    __shared__ __bf16 pS[4][16][32];

    const int tid  = threadIdx.x;
    const int blk  = blockIdx.x;               // b*NH*(T/64) blocks
    const int q64  = blk % (T_ / 64);
    const int bh   = blk / (T_ / 64);
    const int h    = bh % NH_;
    const int b    = bh / NH_;
    const int kvh  = h / (NH_ / NKV_);

    const int wave = tid >> 5;
    const int lane = tid & 31;
    const int half = lane >> 4;
    const int ml   = lane & 15;
    const int m0   = q64 * 64 + wave * 16;

    const __bf16* qp = qb + ((size_t)b * NH_  + h)   * T_ * HD_;
    const __bf16* kp = kb + ((size_t)b * NKV_ + kvh) * T_ * HD_;
    const __bf16* vp = vb + ((size_t)b * NKV_ + kvh) * T_ * HD_;

    // Load the 16x128 query tile as 4 A-fragments (held for the whole pass).
    V16B qfr[4];
#pragma unroll
    for (int cchunk = 0; cchunk < 4; ++cchunk) {
        const size_t rbase = (size_t)(m0 + ml) * HD_ + 32 * cchunk;
#pragma unroll
        for (int e = 0; e < 8; ++e) {
            qfr[cchunk].b[e]     = qp[rbase + 8 * half + e];
            qfr[cchunk].b[e + 8] = qp[rbase + 16 + 8 * half + e];
        }
    }

    V8F y[8];
    float mrow[8], lsum[8];
#pragma unroll
    for (int j = 0; j < 8; ++j)
#pragma unroll
        for (int r = 0; r < 8; ++r) y[j].f[r] = 0.0f;
#pragma unroll
    for (int r = 0; r < 8; ++r) { mrow[r] = -3.0e38f; lsum[r] = 0.0f; }

    const float sc = 0.08838834764831845f;     // 1/sqrt(HD)
    const int nsteps = 2 * q64 + 2;            // (q64*64 + 64)/32, uniform per block

    // Per-thread staging pointers (constant stride per step).
    const int skey = tid >> 2;                 // 0..31
    const int shb  = (tid & 3) * 32;           // hd chunk base
    const __bf16* ksrc = kp + (size_t)skey * HD_ + shb;
    const __bf16* vsrc = vp + (size_t)skey * HD_ + shb;

    for (int st = 0; st < nsteps; ++st) {
        const int kb0 = st * 32;

        // Cooperative stage of K (row-major) and V (transposed) tiles.
        __builtin_prefetch(ksrc + (size_t)32 * HD_, 0, 1);
        __builtin_prefetch(vsrc + (size_t)32 * HD_, 0, 1);
#pragma unroll
        for (int j = 0; j < 32; ++j) Ks[skey][shb + j] = ksrc[j];
#pragma unroll
        for (int j = 0; j < 32; ++j) Vs[shb + j][skey] = vsrc[j];
        ksrc += (size_t)32 * HD_;
        vsrc += (size_t)32 * HD_;
        __syncthreads();

        // scores: two 16x16 fragments (keys kb0..+15 and kb0+16..+31)
        V8F s0, s1;
#pragma unroll
        for (int r = 0; r < 8; ++r) { s0.f[r] = 0.0f; s1.f[r] = 0.0f; }
#pragma unroll
        for (int cchunk = 0; cchunk < 4; ++cchunk) {
            V16B kf0, kf1;
            const int hd = 32 * cchunk + 8 * half;
#pragma unroll
            for (int e = 0; e < 16; ++e) {
                kf0.b[e] = Ks[ml][hd + e + (e >= 8 ? 8 : 0)];
                kf1.b[e] = Ks[16 + ml][hd + e + (e >= 8 ? 8 : 0)];
            }
            s0.v = __builtin_amdgcn_wmma_f32_16x16x32_bf16(
                false, qfr[cchunk].v, false, kf0.v, (short)0, s0.v, false, false);
            s1.v = __builtin_amdgcn_wmma_f32_16x16x32_bf16(
                false, qfr[cchunk].v, false, kf1.v, (short)0, s1.v, false, false);
        }

        // scale + causal mask (soft -inf so exp underflows to 0 cleanly)
#pragma unroll
        for (int r = 0; r < 8; ++r) {
            const int row = m0 + r + 8 * half;
            s0.f[r] = (kb0 + ml      <= row) ? s0.f[r] * sc : -1.0e30f;
            s1.f[r] = (kb0 + 16 + ml <= row) ? s1.f[r] * sc : -1.0e30f;
        }

        // online softmax: row reductions across the 16-lane half
        float alpha[8];
#pragma unroll
        for (int r = 0; r < 8; ++r) {
            float v = fmaxf(s0.f[r], s1.f[r]);
            v = fmaxf(v, __shfl_xor(v, 1, 16));
            v = fmaxf(v, __shfl_xor(v, 2, 16));
            v = fmaxf(v, __shfl_xor(v, 4, 16));
            v = fmaxf(v, __shfl_xor(v, 8, 16));
            const float nm = fmaxf(mrow[r], v);
            alpha[r] = __expf(mrow[r] - nm);
            mrow[r]  = nm;
            s0.f[r] = __expf(s0.f[r] - nm);
            s1.f[r] = __expf(s1.f[r] - nm);
            float rs = s0.f[r] + s1.f[r];
            rs += __shfl_xor(rs, 1, 16);
            rs += __shfl_xor(rs, 2, 16);
            rs += __shfl_xor(rs, 4, 16);
            rs += __shfl_xor(rs, 8, 16);
            lsum[r] = lsum[r] * alpha[r] + rs;
        }
#pragma unroll
        for (int j = 0; j < 8; ++j)
#pragma unroll
            for (int r = 0; r < 8; ++r) y[j].f[r] *= alpha[r];

        // transpose P (16x32) from C/D layout into A layout via LDS
#pragma unroll
        for (int r = 0; r < 8; ++r) {
            pS[wave][r + 8 * half][ml]      = (__bf16)s0.f[r];
            pS[wave][r + 8 * half][16 + ml] = (__bf16)s1.f[r];
        }
        __syncthreads();
        V16B pA;
#pragma unroll
        for (int e = 0; e < 8; ++e) {
            pA.b[e]     = pS[wave][ml][8 * half + e];
            pA.b[e + 8] = pS[wave][ml][16 + 8 * half + e];
        }

        // y(16x128) += P(16x32) @ V(32x128), 8 B-fragments over hd (from Vs)
#pragma unroll
        for (int j = 0; j < 8; ++j) {
            V16B vfr;
#pragma unroll
            for (int e = 0; e < 16; ++e)
                vfr.b[e] = Vs[16 * j + ml][8 * half + e + (e >= 8 ? 8 : 0)];
            y[j].v = __builtin_amdgcn_wmma_f32_16x16x32_bf16(
                false, pA.v, false, vfr.v, (short)0, y[j].v, false, false);
        }
        __syncthreads();   // all waves done reading Ks/Vs/pS before restaging
    }

    // normalize and emit bf16 y in [b, t, h*HD + hd] (feeds the proj GEMM)
    __bf16* ybase = yb + ((size_t)b * T_ + m0 + 8 * half) * C_ + h * HD_ + ml;
#pragma unroll
    for (int j = 0; j < 8; ++j)
#pragma unroll
        for (int r = 0; r < 8; ++r)
            ybase[(size_t)r * C_ + 16 * j] = (__bf16)(y[j].f[r] / lsum[r]);
}

// ---------------------------------------------------------------------------
// Host-side orchestration
// ---------------------------------------------------------------------------
extern "C" void kernel_launch(void* const* d_in, const int* in_sizes, int n_in,
                              void* d_out, int out_size, void* d_ws, size_t ws_size,
                              hipStream_t stream) {
    const float* x     = (const float*)d_in[0];
    const float* ve    = (const float*)d_in[1];
    const float* cosv  = (const float*)d_in[2];
    const float* sinv  = (const float*)d_in[3];
    const float* Wq    = (const float*)d_in[4];
    const float* Wk    = (const float*)d_in[5];
    const float* Wv    = (const float*)d_in[6];
    const float* Wproj = (const float*)d_in[7];
    const float* Wgate = (const float*)d_in[8];
    float* out = (float*)d_out;

    const int M  = B_ * T_;          // 4096
    const int nx = M * C_;           // 8388608  (x, y, out elems)
    const int nq = M * NH_ * HD_;    // 8388608
    const int nk = M * NKV_ * HD_;   // 2097152
    const int nwq = C_ * NH_ * HD_;  // 4194304
    const int nwk = C_ * NKV_ * HD_; // 1048576
    const int nwp = C_ * C_;         // 4194304

    char* ws = (char*)d_ws;
    size_t off = 0;
    auto alloc = [&](size_t bytes) -> void* {
        void* p = ws + off;
        off = (off + bytes + 255) & ~(size_t)255;
        return p;
    };

    __bf16* xb  = (__bf16*)alloc((size_t)nx  * 2);
    __bf16* wqb = (__bf16*)alloc((size_t)nwq * 2);
    __bf16* wkb = (__bf16*)alloc((size_t)nwk * 2);
    __bf16* wvb = (__bf16*)alloc((size_t)nwk * 2);
    __bf16* wpb = (__bf16*)alloc((size_t)nwp * 2);
    float*  qf  = (float*) alloc((size_t)nq  * 4);
    float*  kf  = (float*) alloc((size_t)nk  * 4);
    float*  vf  = (float*) alloc((size_t)nk  * 4);
    __bf16* qbb = (__bf16*)alloc((size_t)nq  * 2);
    __bf16* kbb = (__bf16*)alloc((size_t)nk  * 2);
    __bf16* vbb = (__bf16*)alloc((size_t)nk  * 2);
    __bf16* ybb = (__bf16*)alloc((size_t)nx  * 2);

    auto cvt = [&](const float* src, __bf16* dst, int n) {
        cvt_f32_bf16<<<(n + 255) / 256, 256, 0, stream>>>(src, dst, n);
    };
    cvt(x,     xb,  nx);
    cvt(Wq,    wqb, nwq);
    cvt(Wk,    wkb, nwk);
    cvt(Wv,    wvb, nwk);
    cvt(Wproj, wpb, nwp);

    // QKV projections (K=C=2048; N = 2048 for q, 512 for k/v)
    gemm_bf16_wmma<2048, 2048><<<dim3((NH_ * HD_) / 64, M / 64), 128, 0, stream>>>(
        xb, wqb, qf);
    gemm_bf16_wmma<2048, 512><<<dim3((NKV_ * HD_) / 64, M / 64), 128, 0, stream>>>(
        xb, wkb, kf);
    gemm_bf16_wmma<2048, 512><<<dim3((NKV_ * HD_) / 64, M / 64), 128, 0, stream>>>(
        xb, wvb, vf);

    // gate + ve mix, RoPE + RMS, pack to bf16 [b, head, t, hd]
    postqkv<<<M, 128, 0, stream>>>(x, ve, cosv, sinv, Wgate, qf, kf, vf, qbb, kbb, vbb);

    // causal GQA flash attention
    attn_wmma<<<B_ * NH_ * (T_ / 64), 128, 0, stream>>>(qbb, kbb, vbb, ybb);

    // output projection
    gemm_bf16_wmma<2048, 2048><<<dim3(C_ / 64, M / 64), 128, 0, stream>>>(
        ybb, wpb, out);

    (void)in_sizes; (void)n_in; (void)out_size; (void)ws_size;
}
